// TreeConvolutionNetwork_36326833389618
// MI455X (gfx1250) — matrix operand order repaired and checked by reading, not compiled, not run on hardware
//
#include <hip/hip_runtime.h>

// ---------------------------------------------------------------------------
// TreeConvolutionNetwork for MI455X (gfx1250, wave32, WMMA).
//
// Input flattening assumption (documented):
//   d_in[0] query_vector[512], d_in[1] plan_nodes[65536*384],
//   d_in[2] child_idx[65536*2] (int32), d_in[3] child_mask[65536*2] (f32),
//   then params pytree, dicts flattened in sorted-key order:
//     conv layer i (i=0..2), base=4+6*i: b[64], pb[512], pw[64*512],
//                                        wl[64*512], wp[64*512], wr[64*512]
//     enc layer j (j=0..2), base=22+4*j: W, b, be, g
//         (dims: 512x512, 512x256, 256x128)
//     fin: 34..37 W[512*512],b,be,g ; 38..41 W[512*256],b,be,g ; 42 W[256] ; 43 b[1]
// ---------------------------------------------------------------------------

#define NNODES 65536
#define HIDD   512
#define NFILT  64
#define KSTEP1 48   // 1536 / 32
#define BN     32   // nodes per conv block (2 M-tiles, B-fragment reuse x2)

typedef __attribute__((ext_vector_type(16))) __bf16 v16bf;
typedef __attribute__((ext_vector_type(8)))  float  v8f;

union Frag16 { v16bf v; uint4 q[2]; unsigned short h[16]; };
union Acc8   { v8f   v; float f[8]; };

static __device__ __forceinline__ unsigned short f2bf(float f) {
  union { float f; unsigned u; } v; v.f = f;
  unsigned r = v.u + 0x7FFFu + ((v.u >> 16) & 1u);   // round-to-nearest-even
  return (unsigned short)(r >> 16);
}
static __device__ __forceinline__ float bf2f(unsigned short h) {
  union { unsigned u; float f; } v; v.u = ((unsigned)h) << 16;
  return v.f;
}

// ---------------------------------------------------------------------------
// Weight packing into WMMA B-fragment layout.
// B (32x16 bf16) per wave32: lane l holds column n = l&15; halves i=0..15
// hold K = 16*(l>>4) + i.  Stored as [tileN][kstep][lane][16 halves].
// ---------------------------------------------------------------------------
__global__ __launch_bounds__(256) void pack_w1_kernel(
    const float* __restrict__ wp, const float* __restrict__ wl,
    const float* __restrict__ wr, unsigned short* __restrict__ out) {
  int idx  = blockIdx.x * 256 + threadIdx.x;       // 4*48*32*16 = 98304 total
  int half = idx & 15;
  int lane = (idx >> 4) & 31;
  int rest = idx >> 9;                             // t*48 + s
  int s = rest % KSTEP1;
  int t = rest / KSTEP1;
  int n = t * 16 + (lane & 15);                    // filter (0..63)
  int k = s * 32 + (lane >> 4) * 16 + half;        // 0..1535
  const float* src = (k < 512) ? wp : ((k < 1024) ? wl : wr);
  int kk = k & 511;
  out[idx] = f2bf(src[n * 512 + kk]);              // w[n][k] since y = x @ w.T
}

__global__ __launch_bounds__(256) void pack_w2_kernel(
    const float* __restrict__ pw, unsigned short* __restrict__ out) {
  int idx  = blockIdx.x * 256 + threadIdx.x;       // 32*2*32*16 = 32768 total
  int half = idx & 15;
  int lane = (idx >> 4) & 31;
  int rest = idx >> 9;                             // t*2 + s
  int s = rest & 1;
  int t = rest >> 1;
  int n = t * 16 + (lane & 15);                    // out col (0..511)
  int k = s * 32 + (lane >> 4) * 16 + half;        // 0..63
  out[idx] = f2bf(pw[k * 512 + n]);                // pw[k][n], y = f @ pw
}

// ---------------------------------------------------------------------------
// Query encoder + final MLP helpers (single block, 256 threads, fp32).
// ---------------------------------------------------------------------------
static __device__ void mlp_ln_leaky(const float* __restrict__ W,
                                    const float* __restrict__ b,
                                    const float* __restrict__ g,
                                    const float* __restrict__ be,
                                    int din, int dout,
                                    const float* x, float* y, float* red) {
  int t = threadIdx.x;
  for (int o = t; o < dout; o += 256) {
    float s = b[o];
    for (int k = 0; k < din; ++k) s += x[k] * W[k * dout + o];
    y[o] = s;
  }
  __syncthreads();
  float ls = 0.f;
  for (int o = t; o < dout; o += 256) ls += y[o];
  red[t] = ls; __syncthreads();
  for (int off = 128; off > 0; off >>= 1) { if (t < off) red[t] += red[t + off]; __syncthreads(); }
  float mu = red[0] / (float)dout; __syncthreads();
  float lv = 0.f;
  for (int o = t; o < dout; o += 256) { float d = y[o] - mu; lv += d * d; }
  red[t] = lv; __syncthreads();
  for (int off = 128; off > 0; off >>= 1) { if (t < off) red[t] += red[t + off]; __syncthreads(); }
  float var = red[0] / (float)dout; __syncthreads();
  float rs = rsqrtf(var + 1e-5f);
  for (int o = t; o < dout; o += 256) {
    float v = (y[o] - mu) * rs * g[o] + be[o];
    y[o] = (v >= 0.f) ? v : 0.01f * v;
  }
  __syncthreads();
}

__global__ __launch_bounds__(256) void encode_query_kernel(
    const float* __restrict__ qin,
    const float* W0, const float* b0, const float* be0, const float* g0,
    const float* W1, const float* b1, const float* be1, const float* g1,
    const float* W2, const float* b2, const float* be2, const float* g2,
    float* __restrict__ qout) {
  __shared__ float x[512];
  __shared__ float y[512];
  __shared__ float red[256];
  int t = threadIdx.x;
  x[t] = qin[t]; x[t + 256] = qin[t + 256];
  __syncthreads();
  mlp_ln_leaky(W0, b0, g0, be0, 512, 512, x, y, red);
  mlp_ln_leaky(W1, b1, g1, be1, 512, 256, y, x, red);
  mlp_ln_leaky(W2, b2, g2, be2, 256, 128, x, y, red);
  if (t < 128) qout[t] = y[t];
}

// ---------------------------------------------------------------------------
// Build bf16 augmented activations: X[n] = [plan_nodes[n] (384) | qenc (128)].
// ---------------------------------------------------------------------------
__global__ __launch_bounds__(256) void build_x_kernel(
    const float* __restrict__ plan, const float* __restrict__ qenc,
    unsigned short* __restrict__ Xb) {
  size_t idx = (size_t)blockIdx.x * 256 + threadIdx.x;  // pair index, N*256 total
  int n  = (int)(idx >> 8);
  int cp = (int)(idx & 255) * 2;
  float v0, v1;
  if (cp < 384) { v0 = plan[(size_t)n * 384 + cp]; v1 = plan[(size_t)n * 384 + cp + 1]; }
  else          { v0 = qenc[cp - 384];             v1 = qenc[cp - 383]; }
  unsigned out = (unsigned)f2bf(v0) | ((unsigned)f2bf(v1) << 16);
  ((unsigned*)Xb)[idx] = out;
}

// ---------------------------------------------------------------------------
// Fused tree-conv layer, 32 nodes per block, 4 waves.
// Each wave owns one 16-filter column and BOTH 16-row M-tiles, so every
// B fragment (weights, streamed from L2) feeds two v_wmma ops: halves the
// dominant L2 weight traffic vs one M-tile per wave.
//   Stage 1: resolve per-row gather sources, stage [self|L|R] rows to LDS.
//   Stage 2: f = relu(A[32x1536] @ W[1536x64] + b): 48 k-steps x 2 WMMA.
//   Stage 3: X' = f @ pw + pb: 8 col-tiles x 2 k-steps x 2 M-tiles.
// ---------------------------------------------------------------------------
__global__ __launch_bounds__(128) void conv_layer_kernel(
    const unsigned short* __restrict__ Xin,      // [N][512] bf16
    const int*   __restrict__ child_idx,         // [N][2]
    const float* __restrict__ child_mask,        // [N][2]
    const unsigned short* __restrict__ Bf1,      // [4][48][32][16] bf16
    const float* __restrict__ bconv,             // [64]
    const unsigned short* __restrict__ Bf2,      // [32][2][32][16] bf16
    const float* __restrict__ pb,                // [512]
    unsigned short* __restrict__ Xout) {         // [N][512] bf16
  __shared__ unsigned short As[3 * BN * HIDD];   // 96 KB: self / L / R rows
  __shared__ unsigned short fs[BN * NFILT];      // 4 KB: relu'd filter tile
  __shared__ int rowsrc[3 * BN];                 // row -> source node (-1 = zero)

  const int tid  = threadIdx.x;
  const int base = blockIdx.x * BN;

  // ---- Stage 0: resolve gather sources once per row ----
  if (tid < 3 * BN) {
    int src = tid >> 5;                          // 0=self, 1=L, 2=R
    int m   = tid & 31;
    int node = base + m;
    int s;
    if (src == 0) {
      s = node;
    } else {
      int slot = src - 1;
      float mk = child_mask[node * 2 + slot];
      s = (mk != 0.0f) ? child_idx[node * 2 + slot] : -1;
    }
    rowsrc[tid] = s;
  }
  __syncthreads();

  // ---- Stage 1: cooperative masked gather (16B chunks, coalesced) ----
  for (int i = tid; i < 3 * BN * 64; i += 128) {
    int row = i >> 6;                            // 0..95
    int ch  = i & 63;                            // 16B chunk within 512-half row
    int s = rowsrc[row];
    uint4 val = make_uint4(0u, 0u, 0u, 0u);
    if (s >= 0) val = *(const uint4*)(Xin + (size_t)s * HIDD + ch * 8);
    *(uint4*)(As + row * HIDD + ch * 8) = val;
  }
  __syncthreads();

  const int wv   = tid >> 5;   // wave id 0..3 -> filter tile
  const int lane = tid & 31;
  const int nl   = lane & 15;
  const int kg   = lane >> 4;  // K-half group

  // ---- Stage 2: two f tiles (nodes 0-15, 16-31) for filters [16wv,16wv+16) ----
  Acc8 acc0, acc1;
#pragma unroll
  for (int i = 0; i < 8; ++i) { acc0.f[i] = 0.f; acc1.f[i] = 0.f; }

  const unsigned short* b1p = Bf1 + ((size_t)wv * KSTEP1 * 32 + lane) * 16;
  for (int s = 0; s < KSTEP1; ++s) {
    Frag16 a0, a1, b;
    int kb  = s * 32;
    int src = kb >> 9;                           // which of self/L/R
    int off = kb & 511;
    // A: k order used here (matches packed B) — halves 0..15 at K = 16*kg + i (contiguous)
    const unsigned short* ar0 = As + ((src * BN + nl) * HIDD) + off + kg * 16;
    const unsigned short* ar1 = ar0 + 16 * HIDD;
    a0.q[0] = *(const uint4*)(ar0);
    a0.q[1] = *(const uint4*)(ar0 + 8);
    a1.q[0] = *(const uint4*)(ar1);
    a1.q[1] = *(const uint4*)(ar1 + 8);
    const unsigned short* bp = b1p + (size_t)s * 512;   // 32 lanes * 16 halves
    b.q[0] = *(const uint4*)(bp);
    b.q[1] = *(const uint4*)(bp + 8);
    acc0.v = __builtin_amdgcn_wmma_f32_16x16x32_bf16(
        false, a0.v, false, b.v, (short)0, acc0.v, false, false);
    acc1.v = __builtin_amdgcn_wmma_f32_16x16x32_bf16(
        false, a1.v, false, b.v, (short)0, acc1.v, false, false);
  }
  {
    int n = wv * 16 + nl;
    float bias = bconv[n];
#pragma unroll
    for (int v = 0; v < 8; ++v) {
      int mr = v + kg * 8;                       // C/D layout: M = v + 8*(l>>4)
      float x0 = acc0.f[v] + bias;
      float x1 = acc1.f[v] + bias;
      fs[mr * NFILT + n]        = f2bf(x0 > 0.f ? x0 : 0.f);
      fs[(16 + mr) * NFILT + n] = f2bf(x1 > 0.f ? x1 : 0.f);
    }
  }
  __syncthreads();

  // ---- Stage 3: X' = f @ pw + pb (two M-tiles share each B fragment) ----
  unsigned short* Xo = As;                       // reuse LDS (32x512 bf16)
  Frag16 a00, a01, a10, a11;
  {
    const unsigned short* ar0 = fs + nl * NFILT + kg * 16;
    const unsigned short* ar1 = fs + (16 + nl) * NFILT + kg * 16;
    a00.q[0] = *(const uint4*)(ar0);      a00.q[1] = *(const uint4*)(ar0 + 8);
    a01.q[0] = *(const uint4*)(ar0 + 32); a01.q[1] = *(const uint4*)(ar0 + 40);
    a10.q[0] = *(const uint4*)(ar1);      a10.q[1] = *(const uint4*)(ar1 + 8);
    a11.q[0] = *(const uint4*)(ar1 + 32); a11.q[1] = *(const uint4*)(ar1 + 40);
  }
  for (int tt = 0; tt < 8; ++tt) {
    int t = wv * 8 + tt;                         // out-col tile 0..31
    Acc8 c0, c1;
#pragma unroll
    for (int i = 0; i < 8; ++i) { c0.f[i] = 0.f; c1.f[i] = 0.f; }
    Frag16 b0, b1;
    const unsigned short* bp = Bf2 + ((size_t)t * 1024 + (size_t)lane * 16);
    b0.q[0] = *(const uint4*)(bp);
    b0.q[1] = *(const uint4*)(bp + 8);
    b1.q[0] = *(const uint4*)(bp + 512);
    b1.q[1] = *(const uint4*)(bp + 520);
    c0.v = __builtin_amdgcn_wmma_f32_16x16x32_bf16(
        false, a00.v, false, b0.v, (short)0, c0.v, false, false);
    c0.v = __builtin_amdgcn_wmma_f32_16x16x32_bf16(
        false, a01.v, false, b1.v, (short)0, c0.v, false, false);
    c1.v = __builtin_amdgcn_wmma_f32_16x16x32_bf16(
        false, a10.v, false, b0.v, (short)0, c1.v, false, false);
    c1.v = __builtin_amdgcn_wmma_f32_16x16x32_bf16(
        false, a11.v, false, b1.v, (short)0, c1.v, false, false);
    int n = t * 16 + nl;
    float bo = pb[n];
#pragma unroll
    for (int v = 0; v < 8; ++v) {
      int mr = v + kg * 8;
      Xo[mr * HIDD + n]        = f2bf(c0.f[v] + bo);
      Xo[(16 + mr) * HIDD + n] = f2bf(c1.f[v] + bo);
    }
  }
  __syncthreads();

  for (int i = tid; i < BN * 64; i += 128) {
    int row = i >> 6, ch = i & 63;
    *(uint4*)(Xout + (size_t)(base + row) * HIDD + ch * 8) =
        *(const uint4*)(Xo + row * HIDD + ch * 8);
  }
}

// ---------------------------------------------------------------------------
// Max-pool partials: block b reduces rows [b*512, (b+1)*512), coalesced.
// ---------------------------------------------------------------------------
__global__ __launch_bounds__(256) void maxpool_kernel(
    const unsigned short* __restrict__ X, float* __restrict__ partial) {
  int b = blockIdx.x, t = threadIdx.x;
  float m0 = -3.402823466e38f, m1 = -3.402823466e38f;
  const unsigned* p = (const unsigned*)X + (size_t)b * 512 * 256 + t;
  for (int r = 0; r < 512; ++r) {
    unsigned u = p[(size_t)r * 256];
    m0 = fmaxf(m0, bf2f((unsigned short)(u & 0xFFFFu)));
    m1 = fmaxf(m1, bf2f((unsigned short)(u >> 16)));
  }
  partial[(size_t)b * 512 + 2 * t]     = m0;
  partial[(size_t)b * 512 + 2 * t + 1] = m1;
}

// ---------------------------------------------------------------------------
// Final: reduce partials -> h[512], two LN-leaky layers, final dot -> out[1].
// ---------------------------------------------------------------------------
__global__ __launch_bounds__(256) void final_kernel(
    const float* __restrict__ partial,
    const float* W0, const float* b0, const float* be0, const float* g0,
    const float* W1, const float* b1, const float* be1, const float* g1,
    const float* Wl, const float* bl,
    float* __restrict__ out) {
  __shared__ float x[512];
  __shared__ float y[512];
  __shared__ float red[256];
  int t = threadIdx.x;
  for (int c = t; c < 512; c += 256) {
    float m = -3.402823466e38f;
    for (int r = 0; r < 128; ++r) m = fmaxf(m, partial[(size_t)r * 512 + c]);
    x[c] = m;
  }
  __syncthreads();
  mlp_ln_leaky(W0, b0, g0, be0, 512, 512, x, y, red);
  mlp_ln_leaky(W1, b1, g1, be1, 512, 256, y, x, red);
  red[t] = x[t] * Wl[t];           // din=256: threads handle one term each
  __syncthreads();
  for (int off = 128; off > 0; off >>= 1) { if (t < off) red[t] += red[t + off]; __syncthreads(); }
  if (t == 0) out[0] = red[0] + bl[0];
}

// ---------------------------------------------------------------------------
extern "C" void kernel_launch(void* const* d_in, const int* in_sizes, int n_in,
                              void* d_out, int out_size, void* d_ws, size_t ws_size,
                              hipStream_t stream) {
  (void)in_sizes; (void)n_in; (void)out_size; (void)ws_size;
  const float* query = (const float*)d_in[0];
  const float* plan  = (const float*)d_in[1];
  const int*   cidx  = (const int*)d_in[2];
  const float* cmask = (const float*)d_in[3];

  // conv layer i, base 4+6*i: b, pb, pw, wl, wp, wr
  const float *cb[3], *cpb[3], *cpw[3], *cwl[3], *cwp[3], *cwr[3];
  for (int i = 0; i < 3; ++i) {
    int b = 4 + 6 * i;
    cb[i]  = (const float*)d_in[b + 0];
    cpb[i] = (const float*)d_in[b + 1];
    cpw[i] = (const float*)d_in[b + 2];
    cwl[i] = (const float*)d_in[b + 3];
    cwp[i] = (const float*)d_in[b + 4];
    cwr[i] = (const float*)d_in[b + 5];
  }
  // enc layer j, base 22+4*j: W, b, be, g
  const float *eW[3], *eb[3], *ebe[3], *eg[3];
  for (int j = 0; j < 3; ++j) {
    int b = 22 + 4 * j;
    eW[j]  = (const float*)d_in[b + 0];
    eb[j]  = (const float*)d_in[b + 1];
    ebe[j] = (const float*)d_in[b + 2];
    eg[j]  = (const float*)d_in[b + 3];
  }
  const float* fW0 = (const float*)d_in[34]; const float* fb0 = (const float*)d_in[35];
  const float* fbe0 = (const float*)d_in[36]; const float* fg0 = (const float*)d_in[37];
  const float* fW1 = (const float*)d_in[38]; const float* fb1 = (const float*)d_in[39];
  const float* fbe1 = (const float*)d_in[40]; const float* fg1 = (const float*)d_in[41];
  const float* fWl = (const float*)d_in[42]; const float* fbl = (const float*)d_in[43];

  // Workspace carve-out (256B aligned).
  char* ws = (char*)d_ws;
  size_t off = 0;
  auto carve = [&](size_t bytes) {
    size_t o = off; off += (bytes + 255) & ~(size_t)255; return o;
  };
  float* qenc    = (float*)(ws + carve(128 * sizeof(float)));
  float* partial = (float*)(ws + carve((size_t)128 * 512 * sizeof(float)));
  unsigned short* bf1[3];
  unsigned short* bf2[3];
  for (int i = 0; i < 3; ++i) bf1[i] = (unsigned short*)(ws + carve((size_t)98304 * 2));
  for (int i = 0; i < 3; ++i) bf2[i] = (unsigned short*)(ws + carve((size_t)32768 * 2));
  unsigned short* Xa = (unsigned short*)(ws + carve((size_t)NNODES * HIDD * 2));
  unsigned short* Xb = (unsigned short*)(ws + carve((size_t)NNODES * HIDD * 2));

  // 1) encode query (fp32, single block)
  encode_query_kernel<<<1, 256, 0, stream>>>(
      query,
      eW[0], eb[0], ebe[0], eg[0],
      eW[1], eb[1], ebe[1], eg[1],
      eW[2], eb[2], ebe[2], eg[2],
      qenc);

  // 2) build bf16 augmented X
  build_x_kernel<<<NNODES, 256, 0, stream>>>(plan, qenc, Xa);

  // 3) pack conv weights into WMMA B-fragment layout
  for (int i = 0; i < 3; ++i) {
    pack_w1_kernel<<<384, 256, 0, stream>>>(cwp[i], cwl[i], cwr[i], bf1[i]);
    pack_w2_kernel<<<128, 256, 0, stream>>>(cpw[i], bf2[i]);
  }

  // 4) three fused tree-conv layers (ping-pong Xa <-> Xb)
  conv_layer_kernel<<<NNODES / BN, 128, 0, stream>>>(
      Xa, cidx, cmask, bf1[0], cb[0], bf2[0], cpb[0], Xb);
  conv_layer_kernel<<<NNODES / BN, 128, 0, stream>>>(
      Xb, cidx, cmask, bf1[1], cb[1], bf2[1], cpb[1], Xa);
  conv_layer_kernel<<<NNODES / BN, 128, 0, stream>>>(
      Xa, cidx, cmask, bf1[2], cb[2], bf2[2], cpb[2], Xb);

  // 5) max-pool partials + final MLP
  maxpool_kernel<<<128, 256, 0, stream>>>(Xb, partial);
  final_kernel<<<1, 256, 0, stream>>>(
      partial,
      fW0, fb0, fbe0, fg0,
      fW1, fb1, fbe1, fg1,
      fWl, fbl,
      (float*)d_out);
}
